// QConvImg2Col_62758062129880
// MI455X (gfx1250) — 8-bit matrix operand order repaired and checked by reading, compiled, not run on hardware
//
#include <hip/hip_runtime.h>

typedef __attribute__((ext_vector_type(8))) int v8i;
typedef unsigned int __attribute__((aligned(1))) u32u;

#define CIN   128
#define COUT  128
#define HW    56
#define HWP   58
#define CHP   (HWP * HWP)
#define LSP   (HW * HW)
#define BATCH 32
#define KTOT  1152
#define KSTEP 64
#define KITER (KTOT / KSTEP)
#define NTILE 128
#define NSUB  (NTILE / 16)
#define NTOT  (BATCH * LSP)
#define XQP_BYTES (BATCH * CIN * CHP)
#define WQP_OFF   XQP_BYTES
#define TAB_OFF   (XQP_BYTES + 8 * KITER * 32 * 32)

__device__ __forceinline__ int quant8(float x, float inv_s) {
    float y = x * inv_s;
    y = fminf(fmaxf(y, -128.0f), 127.0f);
    return __float2int_rn(y);
}

__global__ void __launch_bounds__(256) quantize_pad_x(const float* __restrict__ x,
                                                      const float* __restrict__ s_act,
                                                      signed char* __restrict__ xqp) {
    int idx = blockIdx.x * 256 + threadIdx.x;
    float inv_s = 1.0f / s_act[0];
    int bc  = idx / CHP;
    int r   = idx % CHP;
    int ihp = r / HWP;
    int iwp = r % HWP;
    signed char q = 0;
    if (ihp >= 1 && ihp <= HW && iwp >= 1 && iwp <= HW) {
        float v = x[(size_t)bc * LSP + (ihp - 1) * HW + (iwp - 1)];
        q = (signed char)quant8(v, inv_s);
    }
    xqp[idx] = q;
}

__global__ void build_tab(int* __restrict__ tab) {
    int kg = blockIdx.x * 256 + threadIdx.x;
    if (kg < KTOT) {
        int c = kg / 9, t9 = kg % 9;
        tab[kg] = c * CHP + (t9 / 3) * HWP + (t9 % 3);
    }
}

__global__ void pack_w(const float* __restrict__ w,
                       const float* __restrict__ s_w,
                       signed char* __restrict__ wqp) {
    int frag = blockIdx.x;
    int mt = frag / KITER;
    int kt = frag % KITER;
    int lane = threadIdx.x;
    int rowM = lane & 15;
    int half = lane >> 4;
    float inv_s = 1.0f / s_w[0];
    int mm = mt * 16 + rowM;

    unsigned int* dst = (unsigned int*)(wqp + ((size_t)frag * 32 + lane) * 32);
#pragma unroll
    for (int v = 0; v < 8; ++v) {
        int vv = v & 3;
        int kbase = vv * 4 + ((v >= 4) ? 32 : 0) + half * 16;
        unsigned int pk = 0;
#pragma unroll
        for (int j = 0; j < 4; ++j) {
            int kk = kt * KSTEP + kbase + j;
            int q = quant8(w[mm * KTOT + kk], inv_s);
            pk |= (unsigned int)(q & 0xff) << (8 * j);
        }
        dst[v] = pk;
    }
}

__global__ void __launch_bounds__(256) qconv_gemm(const signed char* __restrict__ xqp,
                                                  const signed char* __restrict__ wqp,
                                                  const int* __restrict__ gtab,
                                                  const float* __restrict__ bias,
                                                  const float* __restrict__ s_act,
                                                  const float* __restrict__ s_w,
                                                  float* __restrict__ out) {
    __shared__ __align__(32) signed char lds_b[2][NTILE * KSTEP];
    __shared__ int ltab[KTOT];

    const int tid  = threadIdx.x;
    const int wave = tid >> 5;
    const int lane = tid & 31;
    const int n_base = blockIdx.x * NTILE;
    const float scale = s_act[0] * s_w[0];

    for (int i = tid; i < KTOT; i += 256) ltab[i] = gtab[i];

    const int ng    = tid & 31;
    const int kgrp0 = (tid >> 5) * 2;
    const int N0    = ng * 4;
    const int nb_s  = N0 >> 4;
    const int lane0 = N0 & 15;
    const int n0    = n_base + N0;
    const int b_img = n0 / LSP;
    const int l0    = n0 % LSP;
    const int oh    = l0 / HW;
    const int ow    = l0 % HW;
    const unsigned base_off = (unsigned)(b_img * (CIN * CHP) + oh * HWP + ow);

    int lds_off[2], koff[2];
#pragma unroll
    for (int u = 0; u < 2; ++u) {
        const int kgrp  = kgrp0 + u;
        const int vfrag = (kgrp & 3) + ((kgrp >= 8) ? 4 : 0);
        const int halff = (kgrp >> 2) & 1;
        lds_off[u] = nb_s * 2048 + (halff * 16 + lane0) * 32 + vfrag * 4;
        koff[u]    = kgrp * 4;
    }

    v8i acc[NSUB];
#pragma unroll
    for (int nb = 0; nb < NSUB; ++nb)
#pragma unroll
        for (int r = 0; r < 8; ++r) acc[nb][r] = 0;

    __syncthreads();

    auto stage = [&](int kt, int buf) {
#pragma unroll
        for (int u = 0; u < 2; ++u) {
            const int4 off = *(const int4*)&ltab[kt * KSTEP + koff[u]];
            unsigned int d0 = *(const u32u*)(xqp + (base_off + (unsigned)off.x));
            unsigned int d1 = *(const u32u*)(xqp + (base_off + (unsigned)off.y));
            unsigned int d2 = *(const u32u*)(xqp + (base_off + (unsigned)off.z));
            unsigned int d3 = *(const u32u*)(xqp + (base_off + (unsigned)off.w));
            unsigned int p01 = __builtin_amdgcn_perm(d1, d0, 0x06020400u);
            unsigned int p23 = __builtin_amdgcn_perm(d3, d2, 0x06020400u);
            unsigned int q01 = __builtin_amdgcn_perm(d1, d0, 0x07030501u);
            unsigned int q23 = __builtin_amdgcn_perm(d3, d2, 0x07030501u);
            unsigned int o0 = __builtin_amdgcn_perm(p23, p01, 0x05040100u);
            unsigned int o1 = __builtin_amdgcn_perm(q23, q01, 0x05040100u);
            unsigned int o2 = __builtin_amdgcn_perm(p23, p01, 0x07060302u);
            unsigned int o3 = __builtin_amdgcn_perm(q23, q01, 0x07060302u);
            unsigned int* dst = (unsigned int*)&lds_b[buf][lds_off[u]];
            dst[0]  = o0;
            dst[8]  = o1;
            dst[16] = o2;
            dst[24] = o3;
        }
    };

    auto compute = [&](int kt, int buf) {
        const v8i a = *(const v8i*)(wqp + (((size_t)(wave * KITER + kt)) * 32 + lane) * 32);
#pragma unroll
        for (int nb = 0; nb < NSUB; ++nb) {
            const v8i bf = *(const v8i*)&lds_b[buf][nb * 2048 + lane * 32];
            acc[nb] = __builtin_amdgcn_wmma_i32_16x16x64_iu8(
                 true, a,  true, bf, acc[nb],
                 false,  false);
        }
    };

    stage(0, 0);
    __syncthreads();

    for (int kt = 0; kt < KITER - 1; ++kt) {
        stage(kt + 1, (kt + 1) & 1);
        compute(kt, kt & 1);
        __syncthreads();
    }
    compute(KITER - 1, (KITER - 1) & 1);

    const int half  = lane >> 4;
    const int laneN = lane & 15;
#pragma unroll
    for (int nb = 0; nb < NSUB; ++nb) {
        const int n_tot = n_base + nb * 16 + laneN;
        const int b = n_tot / LSP;
        const int l = n_tot % LSP;
#pragma unroll
        for (int r = 0; r < 8; ++r) {
            const int m = wave * 16 + half * 8 + r;
            out[((size_t)b * COUT + m) * LSP + l] = (float)acc[nb][r] * scale + bias[m];
        }
    }
}

extern "C" void kernel_launch(void* const* d_in, const int* in_sizes, int n_in,
                              void* d_out, int out_size, void* d_ws, size_t ws_size,
                              hipStream_t stream) {
    const float* x     = (const float*)d_in[0];
    const float* w     = (const float*)d_in[1];
    const float* bias  = (const float*)d_in[2];
    const float* s_act = (const float*)d_in[3];
    const float* s_w   = (const float*)d_in[4];
    float* out = (float*)d_out;

    signed char* xqp = (signed char*)d_ws;
    signed char* wqp = (signed char*)d_ws + WQP_OFF;
    int*         tab = (int*)((char*)d_ws + TAB_OFF);

    quantize_pad_x<<<XQP_BYTES / 256, 256, 0, stream>>>(x, s_act, xqp);
    build_tab<<<(KTOT + 255) / 256, 256, 0, stream>>>(tab);
    pack_w<<<8 * KITER, 32, 0, stream>>>(w, s_w, wqp);
    qconv_gemm<<<NTOT / NTILE, 256, 0, stream>>>(xqp, wqp, tab, bias, s_act, s_w, out);
}
